// SelfAttention_44727789420989
// MI455X (gfx1250) — matrix operand order repaired and checked by reading, not compiled, not run on hardware
//
#include <hip/hip_runtime.h>
#include <hip/hip_bf16.h>

// ---------------------------------------------------------------------------
// MI455X (gfx1250) self-attention with prior-gating.
// All GEMMs via v_wmma_f32_16x16x32_bf16 (wave32). fp32 softmax/gating.
// Fused scores->softmax->blend->ctx kernel: probs tile stays in LDS for P@V,
// removing the 256MB global probs re-read (memory-bound roofline ~23us).
// Q tile staged to LDS via TDM tensor_load_to_lds when the builtin exists.
// Gate path split into a 64-block partial reduction to avoid a 4-block
// serial stall on the 16MB hidden_states read.
// ---------------------------------------------------------------------------

#define B_   4
#define S_   1024
#define H_   1024
#define NH_  16
#define HD_  64

typedef __bf16 bf16;
typedef bf16  bf16x4  __attribute__((ext_vector_type(4)));
typedef bf16  bf16x8  __attribute__((ext_vector_type(8)));
typedef bf16  bf16x16 __attribute__((ext_vector_type(16)));
typedef float floatx4 __attribute__((ext_vector_type(4)));
typedef float floatx8 __attribute__((ext_vector_type(8)));
typedef unsigned int uint32x4 __attribute__((ext_vector_type(4)));
typedef int   int32x4 __attribute__((ext_vector_type(4)));
typedef int   int32x8 __attribute__((ext_vector_type(8)));

#if defined(__has_builtin)
#  if __has_builtin(__builtin_amdgcn_tensor_load_to_lds) && \
      __has_builtin(__builtin_amdgcn_s_wait_tensorcnt)
#    define HAVE_TDM 1
#  endif
#endif
#if defined(HAVE_TDM) && defined(__has_include)
#  if __has_include(<hip/amd_detail/amd_gfx1250_TDM.h>)
#    define TDM_SIX_ARGS 1   // therock-10.0 headers -> clang-23 6-arg builtin
#  endif
#endif

static __device__ __forceinline__ floatx8 wmma_bf16(bf16x16 a, bf16x16 b, floatx8 c) {
    return __builtin_amdgcn_wmma_f32_16x16x32_bf16(
        /*neg_a=*/false, a, /*neg_b=*/false, b,
        /*c_mod=*/(short)0, c, /*reuse_a=*/false, /*reuse_b=*/false);
}

// Build a 16-element A/B fragment from two contiguous 8-element chunks.
static __device__ __forceinline__ bf16x16 frag_from2(const bf16* p0, const bf16* p1) {
    bf16x8 lo = *(const bf16x8*)p0;
    bf16x8 hi = *(const bf16x8*)p1;
    bf16x16 a;
#pragma unroll
    for (int i = 0; i < 8; ++i) { a[i] = lo[i]; a[8 + i] = hi[i]; }
    return a;
}

// ---------------------------------------------------------------------------
// Kernel 1: fp32 -> bf16 elementwise convert (hidden states)
// ---------------------------------------------------------------------------
__global__ void k_cvt_x(const float* __restrict__ x, bf16* __restrict__ y) {
    int i = (blockIdx.x * 256 + threadIdx.x) * 4;
    floatx4 v = *(const floatx4*)(x + i);
    bf16x4 o;
#pragma unroll
    for (int j = 0; j < 4; ++j) o[j] = (bf16)v[j];
    *(bf16x4*)(y + i) = o;
}

// ---------------------------------------------------------------------------
// Kernel 2: transpose + convert W (H x H) -> Wt (H x H) bf16, LDS-tiled.
// ---------------------------------------------------------------------------
__global__ void k_wt(const float* __restrict__ W0, const float* __restrict__ W1,
                     const float* __restrict__ W2,
                     bf16* __restrict__ T0, bf16* __restrict__ T1, bf16* __restrict__ T2) {
    const float* W = (blockIdx.z == 0) ? W0 : (blockIdx.z == 1 ? W1 : W2);
    bf16*        T = (blockIdx.z == 0) ? T0 : (blockIdx.z == 1 ? T1 : T2);
    __shared__ float tile[32][33];
    int bx = blockIdx.x * 32, by = blockIdx.y * 32;
    int tx = threadIdx.x, ty = threadIdx.y;   // block (32,8)
#pragma unroll
    for (int j = 0; j < 32; j += 8)
        tile[ty + j][tx] = W[(size_t)(by + ty + j) * H_ + bx + tx];
    __syncthreads();
#pragma unroll
    for (int j = 0; j < 32; j += 8)
        T[(size_t)(bx + ty + j) * H_ + by + tx] = (bf16)tile[tx][ty + j];
}

// ---------------------------------------------------------------------------
// Kernel 3a: gate partial reduction. grid (16, B): each block reduces 64
// sequence rows of masked hidden_states into partial[b][chunk][1024] (+count).
// Deterministic (fixed tree, no float atomics).
// ---------------------------------------------------------------------------
__global__ void k_gate_partial(const float* __restrict__ hs,
                               const float* __restrict__ amask,
                               float* __restrict__ partial,   // [B][16][H]
                               float* __restrict__ pcnt) {    // [B][16]
    int chunk = blockIdx.x;            // 0..15
    int b     = blockIdx.y;
    int t     = threadIdx.x;           // 256 threads, 4 channels each
    int s0    = chunk * 64;
    float a0 = 0.f, a1 = 0.f, a2 = 0.f, a3 = 0.f, cnt = 0.f;
    for (int s = s0; s < s0 + 64; ++s) {
        float m = (amask[b * S_ + s] == 0.f) ? 1.f : 0.f;
        cnt += m;
        if (m != 0.f) {
            const float* row = hs + ((size_t)b * S_ + s) * H_ + t * 4;
            a0 += row[0]; a1 += row[1]; a2 += row[2]; a3 += row[3];
        }
    }
    float* p = partial + ((size_t)b * 16 + chunk) * H_ + t * 4;
    p[0] = a0; p[1] = a1; p[2] = a2; p[3] = a3;
    if (t == 0) pcnt[b * 16 + chunk] = cnt;
}

// ---------------------------------------------------------------------------
// Kernel 3b: gate finalize. One block per batch: combine 16 partials,
// h_ @ Wg, 2-way softmax of logits/2.
// ---------------------------------------------------------------------------
__global__ void k_gate_final(const float* __restrict__ partial,
                             const float* __restrict__ pcnt,
                             const float* __restrict__ Wg, const float* __restrict__ bg,
                             float* __restrict__ gate) {
    int b = blockIdx.x;
    int t = threadIdx.x;               // 256 threads, 4 channels each
    float a0 = 0.f, a1 = 0.f, a2 = 0.f, a3 = 0.f, cnt = 0.f;
#pragma unroll
    for (int c = 0; c < 16; ++c) {
        const float* p = partial + ((size_t)b * 16 + c) * H_ + t * 4;
        a0 += p[0]; a1 += p[1]; a2 += p[2]; a3 += p[3];
        cnt += pcnt[b * 16 + c];
    }
    float inv = 1.f / fmaxf(cnt, 1.f);
    int h = t * 4;
    float l0 = a0 * inv * Wg[(h + 0) * 2 + 0] + a1 * inv * Wg[(h + 1) * 2 + 0] +
               a2 * inv * Wg[(h + 2) * 2 + 0] + a3 * inv * Wg[(h + 3) * 2 + 0];
    float l1 = a0 * inv * Wg[(h + 0) * 2 + 1] + a1 * inv * Wg[(h + 1) * 2 + 1] +
               a2 * inv * Wg[(h + 2) * 2 + 1] + a3 * inv * Wg[(h + 3) * 2 + 1];
    __shared__ float r0[256], r1[256];
    r0[t] = l0; r1[t] = l1;
    __syncthreads();
    for (int off = 128; off > 0; off >>= 1) {
        if (t < off) { r0[t] += r0[t + off]; r1[t] += r1[t + off]; }
        __syncthreads();
    }
    if (t == 0) {
        float a = (r0[0] + bg[0]) * 0.5f;
        float c = (r1[0] + bg[1]) * 0.5f;
        float mx = fmaxf(a, c);
        float e0 = __expf(a - mx), e1 = __expf(c - mx);
        float s = e0 + e1;
        gate[b * 2 + 0] = e0 / s;
        gate[b * 2 + 1] = e1 / s;
    }
}

// ---------------------------------------------------------------------------
// Kernel 4: QKV GEMM. C = X(4096x1024) @ W(1024x1024) + bias, bf16 WMMA.
// grid (32, 16, 3). Q,K stored (b,h,s,d) bf16; V stored transposed (b,h,d,s).
// ---------------------------------------------------------------------------
__global__ void __launch_bounds__(256) k_qkv(
        const bf16* __restrict__ Xbf,
        const bf16* __restrict__ Wt0, const bf16* __restrict__ Wt1, const bf16* __restrict__ Wt2,
        const float* __restrict__ b0, const float* __restrict__ b1, const float* __restrict__ b2,
        bf16* __restrict__ Q, bf16* __restrict__ Kmat, bf16* __restrict__ Vt) {
    int z = blockIdx.z;
    const bf16*  Wt   = (z == 0) ? Wt0 : (z == 1 ? Wt1 : Wt2);
    const float* bias = (z == 0) ? b0  : (z == 1 ? b1  : b2);
    int tid  = threadIdx.x;
    int wave = tid >> 5, lane = tid & 31;
    int g = lane >> 4, lr = lane & 15;
    int m0 = blockIdx.x * 128 + wave * 16;
    int n0 = blockIdx.y * 64;

    floatx8 acc[4] = {};
    const bf16* Arow = Xbf + (size_t)(m0 + lr) * H_;
    for (int k0 = 0; k0 < H_; k0 += 32) {
        bf16x16 a = frag_from2(Arow + k0 + g * 16, Arow + k0 + g * 16 + 8);
#pragma unroll
        for (int s = 0; s < 4; ++s) {
            const bf16* Brow = Wt + (size_t)(n0 + s * 16 + lr) * H_ + k0 + g * 16;
            bf16x16 bfrag = *(const bf16x16*)Brow;   // contiguous 32B of Wt row
            acc[s] = wmma_bf16(a, bfrag, acc[s]);
        }
    }
    int bb  = m0 >> 10;               // batch (tile never straddles batches)
    int sq0 = (m0 & 1023) + g * 8;    // sequence index base
#pragma unroll
    for (int s = 0; s < 4; ++s) {
        int n  = n0 + s * 16 + lr;
        int bh = n >> 6;
        int d  = n & 63;
        float bv = bias[n];
        if (z == 2) {
            bf16x8 pk;                // V^T: 8 consecutive s per lane, packed 16B store
#pragma unroll
            for (int r = 0; r < 8; ++r) pk[r] = (bf16)(acc[s][r] + bv);
            size_t off = ((((size_t)bb * NH_ + bh) * HD_ + d) * S_) + sq0;
            *(bf16x8*)(Vt + off) = pk;
        } else {
            bf16* O = (z == 0) ? Q : Kmat;
#pragma unroll
            for (int r = 0; r < 8; ++r) {
                size_t off = ((((size_t)bb * NH_ + bh) * S_ + sq0 + r) * HD_) + d;
                O[off] = (bf16)(acc[s][r] + bv);
            }
        }
    }
}

// ---------------------------------------------------------------------------
// Kernel 5 (fused): scores = QK^T/8 + mask -> softmax -> blend -> probs,
// then ctx = P @ V with P read back from LDS (no global probs re-read).
// grid (16, NH, B): 64 q-rows per block, k=1024 staged in 256KB dynamic LDS.
// Q tile (64x64 bf16 = 8KB, contiguous) staged via TDM when available.
// ---------------------------------------------------------------------------
__global__ void __launch_bounds__(256) k_attn(
        const bf16* __restrict__ Q, const bf16* __restrict__ Kmat,
        const bf16* __restrict__ Vt,
        const float* __restrict__ amask, const float* __restrict__ prior,
        const float* __restrict__ gate,
        float* __restrict__ probs, float* __restrict__ ctx) {
    __shared__ bf16 qtile[64 * HD_];                      // 8KB static LDS (offset 0)
    extern __shared__ char smem[];
    float* sc     = (float*)smem;                         // [64][1024] fp32
    float* rowmax = (float*)(smem + 64 * 1024 * 4);       // [64]
    float* rowsum = rowmax + 64;                          // [64]
    float* part   = rowsum + 64;                          // [256]

    int b = blockIdx.z, h = blockIdx.y, qt = blockIdx.x;
    int tid = threadIdx.x;
    int wave = tid >> 5, lane = tid & 31;
    int g = lane >> 4, lr = lane & 15;
    int qw = (wave & 3) * 16;           // local q offset (0..48)
    int kh = (wave >> 2) * 512;         // k half

    const bf16* Qbase = Q + (((size_t)b * NH_ + h) * S_ + (size_t)qt * 64) * HD_;

    // ---- Stage Q tile (contiguous 4096 bf16) into LDS ----
#if defined(HAVE_TDM)
    if (tid == 0) {
        // Tensor DMA descriptor (D#), ISA cdna5 ch.8: 1-D tile, 2-byte elements.
        unsigned long long ga = (unsigned long long)(const void*)Qbase;
        uint32x4 g0;
        g0[0] = 1u;                                     // count=1, user mode
        g0[1] = 0u;                                     // lds_addr: qtile at LDS offset 0
        g0[2] = (unsigned)(ga & 0xFFFFFFFFu);           // global_addr[31:0]
        g0[3] = (unsigned)((ga >> 32) & 0x01FFFFFFu) | 0x80000000u; // addr[56:32]|type=2
        int32x8 g1;
        g1[0] = 1 << 16;                                // data_size=1 (2 bytes)
        g1[1] = 4096 << 16;                             // tensor_dim0 = 4096
        g1[2] = 1 << 16;                                // tensor_dim1 = 1
        g1[3] = 4096 << 16;                             // tile_dim0 = 4096
        g1[4] = 1;                                      // tile_dim1 = 1
        g1[5] = 4096;                                   // tensor_dim0_stride = 4096
        g1[6] = 0; g1[7] = 0;
        int32x4 z4 = {0, 0, 0, 0};
#  if defined(TDM_SIX_ARGS)
        int32x8 z8 = {0, 0, 0, 0, 0, 0, 0, 0};
        __builtin_amdgcn_tensor_load_to_lds(g0, g1, z4, z4, z8, 0);
#  else
        __builtin_amdgcn_tensor_load_to_lds(g0, g1, z4, z4, 0);
#  endif
    }
    __builtin_amdgcn_s_wait_tensorcnt(0);               // waves with cnt==0 pass through
    __syncthreads();
#else
    for (int i = tid * 8; i < 64 * HD_; i += 256 * 8)
        *(bf16x8*)(qtile + i) = *(const bf16x8*)(Qbase + i);
    __syncthreads();
#endif

    // A fragments (two d-steps, HD=64), loaded from LDS, reused 32x.
    const bf16* Qrow = qtile + (qw + lr) * HD_;
    bf16x16 a0 = frag_from2(Qrow + g * 16,      Qrow + 8 + g * 16);
    bf16x16 a1 = frag_from2(Qrow + 32 + g * 16, Qrow + 40 + g * 16);

    const bf16* Kbase = Kmat + (((size_t)b * NH_ + h) * S_) * HD_;
    for (int nt = 0; nt < 32; ++nt) {
        int kcol = kh + nt * 16;
        const bf16* Brow = Kbase + (size_t)(kcol + lr) * HD_;
        bf16x16 bf0 = *(const bf16x16*)(Brow + g * 16);
        bf16x16 bf1 = *(const bf16x16*)(Brow + 32 + g * 16);
        floatx8 acc = {};
        acc = wmma_bf16(a0, bf0, acc);
        acc = wmma_bf16(a1, bf1, acc);
        float mval = amask[b * S_ + kcol + lr];   // N fixed per lane
#pragma unroll
        for (int r = 0; r < 8; ++r)
            sc[(qw + g * 8 + r) * 1024 + kcol + lr] = acc[r] * 0.125f + mval;
    }
    __syncthreads();

    // Row max / sum: 4 threads per row, 256 elements each.
    int row = tid >> 2, seg = tid & 3;
    const float* sr = sc + row * 1024 + seg * 256;
    float mx = -3.0e38f;
    for (int i = 0; i < 256; ++i) mx = fmaxf(mx, sr[i]);
    part[tid] = mx;
    __syncthreads();
    if (seg == 0)
        rowmax[row] = fmaxf(fmaxf(part[row * 4], part[row * 4 + 1]),
                            fmaxf(part[row * 4 + 2], part[row * 4 + 3]));
    __syncthreads();
    float rm = rowmax[row];
    float sum = 0.f;
    for (int i = 0; i < 256; ++i) sum += __expf(sr[i] - rm);
    part[tid] = sum;
    __syncthreads();
    if (seg == 0)
        rowsum[row] = part[row * 4] + part[row * 4 + 1] + part[row * 4 + 2] + part[row * 4 + 3];
    __syncthreads();

    // Blend with prior, write probs to global, keep blended P in LDS for P@V.
    float g0f = gate[b * 2 + 0], g1f = gate[b * 2 + 1];
    size_t pbase = (((size_t)b * NH_ + h) * S_ + (size_t)qt * 64) * S_;
    const float* prbase = prior + ((size_t)b * S_ + (size_t)qt * 64) * S_;
    for (int idx = tid; idx < 64 * 1024; idx += 256) {
        int r = idx >> 10;
        __builtin_prefetch(prbase + idx + 2048, 0, 1);  // stream-ahead on prior
        float p = __expf(sc[idx] - rowmax[r]) / rowsum[r];
        float v = p * g0f + prbase[idx] * g1f;
        probs[pbase + idx] = v;
        sc[idx] = v;                                    // own idx only: no race
    }
    __syncthreads();

    // ---- ctx = P(64x1024, LDS fp32->bf16) @ V(1024x64, bf16 via V^T rows) ----
    int qw2 = (wave & 3) * 16;          // wave tile: 16(M) x 32(N)
    int dh  = (wave >> 2) * 32;
    floatx8 cacc[2] = {};
    const bf16* Vbase = Vt + (((size_t)b * NH_ + h) * HD_ + dh) * S_;
    const float* Pr = sc + (qw2 + lr) * 1024;
    for (int k0 = 0; k0 < S_; k0 += 32) {
        const float* pr = Pr + k0;
        floatx4 f0 = *(const floatx4*)(pr + g * 16);
        floatx4 f1 = *(const floatx4*)(pr + g * 16 + 4);
        floatx4 f2 = *(const floatx4*)(pr + g * 16 + 8);
        floatx4 f3 = *(const floatx4*)(pr + g * 16 + 12);
        bf16x16 a;
#pragma unroll
        for (int i = 0; i < 4; ++i) {
            a[i]      = (bf16)f0[i];
            a[4 + i]  = (bf16)f1[i];
            a[8 + i]  = (bf16)f2[i];
            a[12 + i] = (bf16)f3[i];
        }
#pragma unroll
        for (int s = 0; s < 2; ++s) {
            const bf16* vr = Vbase + (size_t)(s * 16 + lr) * S_ + k0 + g * 16;
            bf16x16 bfrag = *(const bf16x16*)vr;
            cacc[s] = wmma_bf16(a, bfrag, cacc[s]);
        }
    }
#pragma unroll
    for (int s = 0; s < 2; ++s) {
        int d = dh + s * 16 + lr;
#pragma unroll
        for (int r = 0; r < 8; ++r) {
            int q = qt * 64 + qw2 + g * 8 + r;
            ctx[((size_t)b * S_ + q) * H_ + h * HD_ + d] = cacc[s][r];
        }
    }
}

// ---------------------------------------------------------------------------
extern "C" void kernel_launch(void* const* d_in, const int* in_sizes, int n_in,
                              void* d_out, int out_size, void* d_ws, size_t ws_size,
                              hipStream_t stream) {
    (void)in_sizes; (void)n_in; (void)out_size; (void)ws_size;
    const float* hs    = (const float*)d_in[0];
    const float* amask = (const float*)d_in[1];
    const float* prior = (const float*)d_in[2];
    const float* Wq    = (const float*)d_in[3];
    const float* bq    = (const float*)d_in[4];
    const float* Wk    = (const float*)d_in[5];
    const float* bk    = (const float*)d_in[6];
    const float* Wv    = (const float*)d_in[7];
    const float* bv    = (const float*)d_in[8];
    const float* Wg    = (const float*)d_in[9];
    const float* bg    = (const float*)d_in[10];

    float* ctx_out   = (float*)d_out;                       // (B,S,H)
    float* probs_out = ctx_out + (size_t)B_ * S_ * H_;      // (B,NH,S,S)

    // Workspace layout (bf16 staging + gate partials, ~40MB)
    char* w = (char*)d_ws;
    bf16* Xbf  = (bf16*)w; w += (size_t)B_ * S_ * H_ * 2;
    bf16* Wtq  = (bf16*)w; w += (size_t)H_ * H_ * 2;
    bf16* Wtk  = (bf16*)w; w += (size_t)H_ * H_ * 2;
    bf16* Wtv  = (bf16*)w; w += (size_t)H_ * H_ * 2;
    bf16* Qb   = (bf16*)w; w += (size_t)B_ * NH_ * S_ * HD_ * 2;
    bf16* Kb   = (bf16*)w; w += (size_t)B_ * NH_ * S_ * HD_ * 2;
    bf16* Vtb  = (bf16*)w; w += (size_t)B_ * NH_ * HD_ * S_ * 2;
    float* gpart = (float*)w; w += (size_t)B_ * 16 * H_ * 4;
    float* gcnt  = (float*)w; w += (size_t)B_ * 16 * 4;
    float* gate  = (float*)w;

    k_cvt_x<<<(B_ * S_ * H_) / 1024, 256, 0, stream>>>(hs, Xbf);
    k_wt<<<dim3(32, 32, 3), dim3(32, 8), 0, stream>>>(Wq, Wk, Wv, Wtq, Wtk, Wtv);
    k_gate_partial<<<dim3(16, B_), 256, 0, stream>>>(hs, amask, gpart, gcnt);
    k_gate_final<<<B_, 256, 0, stream>>>(gpart, gcnt, Wg, bg, gate);
    k_qkv<<<dim3(32, 16, 3), 256, 0, stream>>>(Xbf, Wtq, Wtk, Wtv, bq, bk, bv, Qb, Kb, Vtb);
    size_t smem = (size_t)64 * 1024 * 4 + 64 * 4 + 64 * 4 + 256 * 4;
    k_attn<<<dim3(16, NH_, B_), 256, smem, stream>>>(Qb, Kb, Vtb, amask, prior, gate,
                                                     probs_out, ctx_out);
}
